// ResponseDecoder_84482006712444
// MI455X (gfx1250) — matrix operand order repaired and checked by reading, not compiled, not run on hardware
//
#include <hip/hip_runtime.h>

typedef __attribute__((ext_vector_type(16))) __bf16 v16bf;
typedef __attribute__((ext_vector_type(8)))  __bf16 v8bf;
typedef __attribute__((ext_vector_type(8)))  float  v8f;

#define TPB 256

// ---------------- conversion kernels ----------------
__global__ void k_conv_bf16(const float* __restrict__ in, __bf16* __restrict__ out, int n) {
    int i = blockIdx.x * TPB + threadIdx.x;
    if (i < n) out[i] = (__bf16)in[i];
}

// in is (R rows, C cols) f32 row-major; out is (C, R) bf16 row-major (i.e. transpose)
__global__ void k_convT_bf16(const float* __restrict__ in, __bf16* __restrict__ out, int R, int C) {
    int i = blockIdx.x * TPB + threadIdx.x;
    if (i < R * C) {
        int r = i / C, c = i % C;
        out[(size_t)c * R + r] = (__bf16)in[i];
    }
}

// ---------------- embedding gather into x slice ----------------
__global__ void k_embed(const float* __restrict__ emb, const int* __restrict__ ids,
                        float* __restrict__ x, int E, int xld, int col0, int n) {
    int i = blockIdx.x * TPB + threadIdx.x;
    if (i < n) {
        int b = i / E, e = i % E;
        x[(size_t)b * xld + col0 + e] = emb[(size_t)ids[b] * E + e];
    }
}

// ---------------- WMMA bf16 GEMM: C(MxN) = A(MxK) @ B(KxN) [+ bias(N)] ----------------
// 256 threads = 8 waves; block computes a 64x128 C tile; each wave a 32x32 quadrant
// (2 A-frags x 2 B-frags = 4 v_wmma per k-step). Requires M%64==0, N%128==0, K%32==0.
//
// A tile stored in LDS with k contiguous per row (no swizzle) so that each
// lane's 16-element A fragment (k = 16*(lane/16) + s: [0..15] / [16..31], same as B) is a
// contiguous 32B read. B tile stored transposed BsT[n][k] so each lane's B fragment
// (k contiguous per lane) is also a contiguous 32B read.
__global__ void k_gemm_bf16(const __bf16* __restrict__ A, const __bf16* __restrict__ Bm,
                            const float* __restrict__ bias, float* __restrict__ C,
                            int M, int N, int K) {
    const int LDA = 40;   // 80B row stride: 16B-aligned, conflict-breaking
    const int LDB = 40;
    __shared__ __align__(32) __bf16 As[64 * 40];     // 64 rows x 32 k (k contiguous)
    __shared__ __align__(32) __bf16 BsT[128 * 40];   // 128 n x 32 k

    const int tid  = threadIdx.x;
    const int lane = tid & 31;
    const int wave = tid >> 5;       // 0..7
    const int wm   = wave >> 2;      // 0..1  (M quadrant of 32 rows)
    const int wn   = wave & 3;       // 0..3  (N quadrant of 32 cols)
    const int m0   = blockIdx.y * 64;
    const int n0   = blockIdx.x * 128;

    // cooperative-load assignments
    const int arow = tid >> 2;            // 0..63
    const int ac0  = (tid & 3) * 8;       // 0,8,16,24
    const int asc  = ac0;   // col base (k contiguous)
    const int bn   = tid & 127;           // 0..127
    const int bkg  = (tid >> 7) * 16;     // 0 or 16

    const int lm = lane & 15;
    const int lk = (lane >> 4) * 16;

    v8f zero = {};
    v8f acc[2][2];
#pragma unroll
    for (int i = 0; i < 2; ++i)
#pragma unroll
        for (int j = 0; j < 2; ++j) acc[i][j] = zero;

    for (int k0 = 0; k0 < K; k0 += 32) {
        // A tile 64x32 -> LDS (one b128 global load + one b128 LDS store per thread)
        *(v8bf*)&As[arow * LDA + asc] =
            *(const v8bf*)&A[(size_t)(m0 + arow) * K + (k0 + ac0)];

        // B tile 32x128 -> LDS transposed (coalesced u16 global reads per k row,
        // vectorized 16B LDS stores)
        v8bf b0, b1;
#pragma unroll
        for (int j = 0; j < 8; ++j) {
            b0[j] = Bm[(size_t)(k0 + bkg + j) * N + (n0 + bn)];
            b1[j] = Bm[(size_t)(k0 + bkg + 8 + j) * N + (n0 + bn)];
        }
        *(v8bf*)&BsT[bn * LDB + bkg]     = b0;
        *(v8bf*)&BsT[bn * LDB + bkg + 8] = b1;
        __syncthreads();

        // fragment loads: contiguous 32B per lane -> 2x ds_load_b128 each
        v16bf afr[2], bfr[2];
#pragma unroll
        for (int i = 0; i < 2; ++i) {
            const __bf16* pa = &As[(wm * 32 + i * 16 + lm) * LDA + lk];
            afr[i] = __builtin_shufflevector(*(const v8bf*)pa, *(const v8bf*)(pa + 8),
                                             0, 1, 2, 3, 4, 5, 6, 7,
                                             8, 9, 10, 11, 12, 13, 14, 15);
            const __bf16* pb = &BsT[(wn * 32 + i * 16 + lm) * LDB + lk];
            bfr[i] = __builtin_shufflevector(*(const v8bf*)pb, *(const v8bf*)(pb + 8),
                                             0, 1, 2, 3, 4, 5, 6, 7,
                                             8, 9, 10, 11, 12, 13, 14, 15);
        }

#pragma unroll
        for (int i = 0; i < 2; ++i)
#pragma unroll
            for (int j = 0; j < 2; ++j)
                acc[i][j] = __builtin_amdgcn_wmma_f32_16x16x32_bf16(
                                false, afr[i], false, bfr[j], (short)0, acc[i][j],
                                false, false);
        __syncthreads();
    }

    // C/D layout: VGPR r holds M = r (+8 for lanes 16..31), N = lane&15
    const int mb = m0 + wm * 32 + ((lane >> 4) * 8);
    const int nb = n0 + wn * 32 + lm;
#pragma unroll
    for (int i = 0; i < 2; ++i) {
#pragma unroll
        for (int j = 0; j < 2; ++j) {
            const int n_g = nb + j * 16;
            const float bv = bias ? bias[n_g] : 0.0f;
#pragma unroll
            for (int r = 0; r < 8; ++r)
                C[(size_t)(mb + i * 16 + r) * N + n_g] = acc[i][j][r] + bv;
        }
    }
}

// ---------------- e[b,t] = sum_h tanh(encW[t*B+b,h] + hW[b,h]) * v[h]  (+ *post) ----------------
__global__ void k_energy_dot(const float* __restrict__ encW, const float* __restrict__ hW,
                             const float* __restrict__ v, const float* __restrict__ post,
                             float* __restrict__ e_out, int T, int B, int H) {
    __shared__ float red[TPB];
    const int idx = blockIdx.x;          // = t*B + b
    const int b = idx % B, t = idx / B;
    const int tid = threadIdx.x;

    float p = 0.0f;
    for (int h = tid; h < H; h += TPB)
        p += tanhf(encW[(size_t)idx * H + h] + hW[(size_t)b * H + h]) * v[h];
    red[tid] = p;
    __syncthreads();
    for (int s = TPB / 2; s > 0; s >>= 1) {
        if (tid < s) red[tid] += red[tid + s];
        __syncthreads();
    }
    if (tid == 0) e_out[(size_t)b * T + t] = red[0] + (post ? *post : 0.0f);
}

// ---------------- w = softmax_t(e[b,:]); x[b, col0+h] = sum_t w[t] * enc[t,b,h] ----------------
__global__ void k_softmax_ctx(const float* __restrict__ e, const float* __restrict__ enc,
                              float* __restrict__ xout, int T, int B, int H,
                              int xld, int col0) {
    __shared__ float w[TPB];
    __shared__ float red[TPB];
    const int b = blockIdx.x;
    const int tid = threadIdx.x;

    float val = (tid < T) ? e[(size_t)b * T + tid] : -1e30f;
    red[tid] = val;
    __syncthreads();
    for (int s = TPB / 2; s > 0; s >>= 1) {
        if (tid < s) red[tid] = fmaxf(red[tid], red[tid + s]);
        __syncthreads();
    }
    const float mx = red[0];
    __syncthreads();

    float ex = (tid < T) ? expf(val - mx) : 0.0f;
    w[tid] = ex;
    red[tid] = ex;
    __syncthreads();
    for (int s = TPB / 2; s > 0; s >>= 1) {
        if (tid < s) red[tid] += red[tid + s];
        __syncthreads();
    }
    const float S = red[0];
    __syncthreads();
    w[tid] = w[tid] / S;
    __syncthreads();

    for (int h = tid; h < H; h += TPB) {
        float acc = 0.0f;
        for (int t = 0; t < T; ++t)
            acc += w[t] * enc[((size_t)t * B + b) * H + h];
        xout[(size_t)b * xld + col0 + h] = acc;
    }
}

// ---------------- GRU cell + LayerNorm (ddof=1) ----------------
__global__ void k_gru_ln(const float* __restrict__ gx, const float* __restrict__ gh,
                         const float* __restrict__ h0, const float* __restrict__ ln_a,
                         const float* __restrict__ ln_b, float* __restrict__ h_new_out,
                         __bf16* __restrict__ gru_bf, int H) {
    __shared__ float hbuf[1024];
    __shared__ float red[TPB];
    const int b = blockIdx.x;
    const int tid = threadIdx.x;
    const int H3 = 3 * H;

    for (int i = tid; i < H; i += TPB) {
        float r  = 1.0f / (1.0f + expf(-(gx[(size_t)b * H3 + i]         + gh[(size_t)b * H3 + i])));
        float zg = 1.0f / (1.0f + expf(-(gx[(size_t)b * H3 + H + i]     + gh[(size_t)b * H3 + H + i])));
        float ng = tanhf(gx[(size_t)b * H3 + 2 * H + i] + r * gh[(size_t)b * H3 + 2 * H + i]);
        float h  = (1.0f - zg) * ng + zg * h0[(size_t)b * H + i];
        hbuf[i] = h;
        h_new_out[(size_t)b * H + i] = h;
    }
    __syncthreads();

    float s1 = 0.0f, s2 = 0.0f;
    for (int i = tid; i < H; i += TPB) { float h = hbuf[i]; s1 += h; s2 += h * h; }
    red[tid] = s1;
    __syncthreads();
    for (int s = TPB / 2; s > 0; s >>= 1) { if (tid < s) red[tid] += red[tid + s]; __syncthreads(); }
    const float sum = red[0];
    __syncthreads();
    red[tid] = s2;
    __syncthreads();
    for (int s = TPB / 2; s > 0; s >>= 1) { if (tid < s) red[tid] += red[tid + s]; __syncthreads(); }
    const float sumsq = red[0];
    __syncthreads();

    const float mean = sum / (float)H;
    float var = (sumsq - sum * mean) / (float)(H - 1);
    var = var > 0.0f ? var : 0.0f;
    const float inv = 1.0f / (sqrtf(var) + 0.001f);

    for (int i = tid; i < H; i += TPB) {
        float g = (hbuf[i] - mean) * inv * ln_a[i] + ln_b[i];
        gru_bf[(size_t)b * H + i] = (__bf16)g;
    }
}

// ---------------- joint softmax over [gen(V); zc(T)] + copy mix ----------------
__global__ void k_mix(const float* __restrict__ gen, const float* __restrict__ zc,
                      const float* __restrict__ pz, float* __restrict__ out,
                      int T, int B, int V, int freq) {
    __shared__ float red[TPB];
    __shared__ float cw[TPB];
    const int b = blockIdx.x;
    const int tid = threadIdx.x;

    // max over V + T
    float m = -1e30f;
    for (int i = tid; i < V; i += TPB) m = fmaxf(m, gen[(size_t)b * V + i]);
    if (tid < T) m = fmaxf(m, zc[(size_t)b * T + tid]);
    red[tid] = m;
    __syncthreads();
    for (int s = TPB / 2; s > 0; s >>= 1) { if (tid < s) red[tid] = fmaxf(red[tid], red[tid + s]); __syncthreads(); }
    const float M = red[0];
    __syncthreads();

    // sum of exps
    float s = 0.0f;
    for (int i = tid; i < V; i += TPB) s += expf(gen[(size_t)b * V + i] - M);
    if (tid < T) s += expf(zc[(size_t)b * T + tid] - M);
    red[tid] = s;
    __syncthreads();
    for (int st = TPB / 2; st > 0; st >>= 1) { if (tid < st) red[tid] += red[tid + st]; __syncthreads(); }
    const float S = red[0];
    __syncthreads();

    if (tid < T) cw[tid] = expf(zc[(size_t)b * T + tid] - M) / S;
    __syncthreads();

    const float invS = 1.0f / S;
    for (int v = tid; v < V; v += TPB) {
        float p = expf(gen[(size_t)b * V + v] - M) * invS;
        if (v >= freq) {
            float acc = 0.0f;
            for (int t = 0; t < T; ++t)
                acc += cw[t] * pz[((size_t)t * B + b) * (size_t)V + v];
            p += acc;
        }
        out[(size_t)b * V + v] = p;
    }
}

// ==================================================================================
extern "C" void kernel_launch(void* const* d_in, const int* in_sizes, int n_in,
                              void* d_out, int out_size, void* d_ws, size_t ws_size,
                              hipStream_t stream) {
    const int E = 512, H = 1024, V = 32000, B = 64, TZ = 64, TU = 256, FREQ = 500;
    const int XK = E + 2 * H;   // 2560

    const float* z_enc  = (const float*)d_in[0];
    const float* pz     = (const float*)d_in[1];
    const float* u_enc  = (const float*)d_in[2];
    const int*   m_t    = (const int*)d_in[3];
    const float* last_h = (const float*)d_in[4];
    const float* emb    = (const float*)d_in[5];
    const float* azW = (const float*)d_in[6];
    const float* azb = (const float*)d_in[7];
    const float* azv = (const float*)d_in[8];
    const float* auW = (const float*)d_in[9];
    const float* aub = (const float*)d_in[10];
    const float* auv = (const float*)d_in[11];
    const float* Wih = (const float*)d_in[12];
    const float* Whh = (const float*)d_in[13];
    const float* bih = (const float*)d_in[14];
    const float* bhh = (const float*)d_in[15];
    const float* lna = (const float*)d_in[16];
    const float* lnb = (const float*)d_in[17];
    const float* pW  = (const float*)d_in[18];
    const float* pb  = (const float*)d_in[19];
    const float* cW  = (const float*)d_in[20];
    const float* cb  = (const float*)d_in[21];
    const float* v1W = (const float*)d_in[22];
    const float* v1b = (const float*)d_in[23];

    float* proba = (float*)d_out;                       // (B, V)
    float* h_new = (float*)d_out + (size_t)B * V;       // (1, B, H)

    // ---- workspace bump allocator ----
    size_t off = 0;
    auto alloc = [&](size_t bytes) -> void* {
        void* p = (char*)d_ws + off;
        off += (bytes + 255) & ~(size_t)255;
        return p;
    };

    __bf16* z_bf   = (__bf16*)alloc((size_t)TZ * B * H * 2);
    __bf16* u_bf   = (__bf16*)alloc((size_t)TU * B * H * 2);
    __bf16* azW_bf = (__bf16*)alloc((size_t)2 * H * H * 2);
    __bf16* auW_bf = (__bf16*)alloc((size_t)2 * H * H * 2);
    __bf16* cW_bf  = (__bf16*)alloc((size_t)2 * H * H * 2);
    __bf16* pW_bf  = (__bf16*)alloc((size_t)H * V * 2);
    __bf16* ihT_bf = (__bf16*)alloc((size_t)XK * 3 * H * 2);
    __bf16* hhT_bf = (__bf16*)alloc((size_t)H * 3 * H * 2);
    __bf16* h0_bf  = (__bf16*)alloc((size_t)B * H * 2);
    __bf16* x_bf   = (__bf16*)alloc((size_t)B * XK * 2);
    __bf16* gru_bf = (__bf16*)alloc((size_t)B * H * 2);

    float* hWz   = (float*)alloc((size_t)B * H * 4);
    float* hWu   = (float*)alloc((size_t)B * H * 4);
    float* eWz   = (float*)alloc((size_t)TZ * B * H * 4);
    float* eWu   = (float*)alloc((size_t)TU * B * H * 4);
    float* e_z   = (float*)alloc((size_t)B * TZ * 4);
    float* e_u   = (float*)alloc((size_t)B * TU * 4);
    float* x_f32 = (float*)alloc((size_t)B * XK * 4);
    float* gx    = (float*)alloc((size_t)B * 3 * H * 4);
    float* gh    = (float*)alloc((size_t)B * 3 * H * 4);
    float* g2    = (float*)alloc((size_t)B * H * 4);
    float* e2    = (float*)alloc((size_t)TZ * B * H * 4);
    float* zcopy = (float*)alloc((size_t)B * TZ * 4);
    float* gen   = (float*)alloc((size_t)B * V * 4);

    auto cdiv = [](int a, int b) { return (a + b - 1) / b; };
    int n;

    // ---- precision conversion (once per launch) ----
    n = TZ * B * H;  k_conv_bf16<<<cdiv(n, TPB), TPB, 0, stream>>>(z_enc, z_bf, n);
    n = TU * B * H;  k_conv_bf16<<<cdiv(n, TPB), TPB, 0, stream>>>(u_enc, u_bf, n);
    n = 2 * H * H;   k_conv_bf16<<<cdiv(n, TPB), TPB, 0, stream>>>(azW, azW_bf, n);
    n = 2 * H * H;   k_conv_bf16<<<cdiv(n, TPB), TPB, 0, stream>>>(auW, auW_bf, n);
    n = 2 * H * H;   k_conv_bf16<<<cdiv(n, TPB), TPB, 0, stream>>>(cW,  cW_bf,  n);
    n = H * V;       k_conv_bf16<<<cdiv(n, TPB), TPB, 0, stream>>>(pW,  pW_bf,  n);
    n = 3 * H * XK;  k_convT_bf16<<<cdiv(n, TPB), TPB, 0, stream>>>(Wih, ihT_bf, 3 * H, XK);
    n = 3 * H * H;   k_convT_bf16<<<cdiv(n, TPB), TPB, 0, stream>>>(Whh, hhT_bf, 3 * H, H);
    n = B * H;       k_conv_bf16<<<cdiv(n, TPB), TPB, 0, stream>>>(last_h, h0_bf, n);

    // ---- embedding gather into x[:, 2H:2H+E] ----
    n = B * E;
    k_embed<<<cdiv(n, TPB), TPB, 0, stream>>>(emb, m_t, x_f32, E, XK, 2 * H, n);

    // GEMM launch geometry: block 256, grid (N/128, M/64)
    // ---- attention over z_enc ----
    k_gemm_bf16<<<dim3(H / 128, B / 64), TPB, 0, stream>>>(h0_bf, azW_bf, azb, hWz, B, H, H);
    k_gemm_bf16<<<dim3(H / 128, (TZ * B) / 64), TPB, 0, stream>>>(z_bf, azW_bf + (size_t)H * H, nullptr, eWz, TZ * B, H, H);
    k_energy_dot<<<TZ * B, TPB, 0, stream>>>(eWz, hWz, azv, nullptr, e_z, TZ, B, H);
    k_softmax_ctx<<<B, TPB, 0, stream>>>(e_z, z_enc, x_f32, TZ, B, H, XK, 0);

    // ---- attention over u_enc ----
    k_gemm_bf16<<<dim3(H / 128, B / 64), TPB, 0, stream>>>(h0_bf, auW_bf, aub, hWu, B, H, H);
    k_gemm_bf16<<<dim3(H / 128, (TU * B) / 64), TPB, 0, stream>>>(u_bf, auW_bf + (size_t)H * H, nullptr, eWu, TU * B, H, H);
    k_energy_dot<<<TU * B, TPB, 0, stream>>>(eWu, hWu, auv, nullptr, e_u, TU, B, H);
    k_softmax_ctx<<<B, TPB, 0, stream>>>(e_u, u_enc, x_f32, TU, B, H, XK, H);

    // ---- GRU input x (bf16) and gates ----
    n = B * XK;  k_conv_bf16<<<cdiv(n, TPB), TPB, 0, stream>>>(x_f32, x_bf, n);
    k_gemm_bf16<<<dim3(3 * H / 128, B / 64), TPB, 0, stream>>>(x_bf,  ihT_bf, bih, gx, B, 3 * H, XK);
    k_gemm_bf16<<<dim3(3 * H / 128, B / 64), TPB, 0, stream>>>(h0_bf, hhT_bf, bhh, gh, B, 3 * H, H);
    k_gru_ln<<<B, TPB, 0, stream>>>(gx, gh, last_h, lna, lnb, h_new, gru_bf, H);

    // ---- generation scores: gru_out @ proj_W + proj_b ----
    k_gemm_bf16<<<dim3(V / 128, B / 64), TPB, 0, stream>>>(gru_bf, pW_bf, pb, gen, B, V, H);

    // ---- copy scores: tanh([z_enc; gru] @ copy_W + copy_b) . v1_W + v1_b ----
    k_gemm_bf16<<<dim3(H / 128, B / 64), TPB, 0, stream>>>(gru_bf, cW_bf + (size_t)H * H, cb, g2, B, H, H);
    k_gemm_bf16<<<dim3(H / 128, (TZ * B) / 64), TPB, 0, stream>>>(z_bf, cW_bf, nullptr, e2, TZ * B, H, H);
    k_energy_dot<<<TZ * B, TPB, 0, stream>>>(e2, g2, v1W, v1b, zcopy, TZ, B, H);

    // ---- joint softmax + copy mix (HBM-bound: streams the 524MB pz_proba once) ----
    k_mix<<<B, TPB, 0, stream>>>(gen, zcopy, pz, proba, TZ, B, V, FREQ);
}
